// Attention_20624432956082
// MI455X (gfx1250) — matrix operand order repaired and checked by reading, not compiled, not run on hardware
//
#include <hip/hip_runtime.h>
#include <math.h>

// ---------------- problem constants ----------------
#define B_     1024
#define T_     32
#define IN_    256
#define H_     256
#define C_     38
#define STEPS_ 26
#define KX_    294            // w_ih row stride = IN + C

typedef __attribute__((ext_vector_type(16))) __bf16 v16bf;
typedef __attribute__((ext_vector_type(8)))  float  v8f;

union FragU { v16bf v; uint4 q[2]; };

__device__ __forceinline__ unsigned short f2bf(float f) {
  unsigned u = __float_as_uint(f);
  unsigned r = u + 0x7fffu + ((u >> 16) & 1u);   // round-to-nearest-even
  return (unsigned short)(r >> 16);
}

__device__ __forceinline__ float sigm(float x) { return 1.0f / (1.0f + __expf(-x)); }

// A-matrix 16x32 bf16 fragment (rows = M, row-major source, stride 256 bf16).
// lanes 0-15: row=lane, K=kb..kb+15
// lanes 16-31: row=lane-16, K=kb+16..kb+31
__device__ __forceinline__ v16bf load_frag_A(const unsigned short* base, int lane, int kb) {
  int row = lane & 15, hi = lane >> 4;
  const uint4* p = (const uint4*)(base + (size_t)row * 256 + kb + hi * 16);
  FragU f; f.q[0] = p[0]; f.q[1] = p[1];
  return f.v;
}

// B-matrix 32x16 bf16 fragment from W stored row-major (Nrows, 256) with
// element B[k][n] = W[ntile*16+n][kb+k]. lanes 0-15: K=kb..kb+15 at N=lane,
// lanes 16-31: K=kb+16..kb+31 at N=lane-16. 16 contiguous bf16 per lane.
__device__ __forceinline__ v16bf load_frag_B(const unsigned short* W, int lane, int ntile, int kb) {
  int col = lane & 15, hi = lane >> 4;
  const uint4* p = (const uint4*)(W + (size_t)(ntile * 16 + col) * 256 + kb + hi * 16);
  FragU f; f.q[0] = p[0]; f.q[1] = p[1];
  return f.v;
}

__device__ __forceinline__ v8f wmma_bf16(v16bf a, v16bf b, v8f c) {
  return __builtin_amdgcn_wmma_f32_16x16x32_bf16(false, a, false, b, (short)0, c, false, false);
}

__device__ __forceinline__ v8f v8zero() {
  v8f z = {0.f, 0.f, 0.f, 0.f, 0.f, 0.f, 0.f, 0.f};
  return z;
}

// ---------------- prep kernels ----------------

// bf16 copies of weights + combined bias. grid: 1024 blocks x 256.
__global__ void conv_weights(const float* __restrict__ w_ih, const float* __restrict__ w_hh,
                             const float* __restrict__ w_h2h, const float* __restrict__ w_i2h,
                             const float* __restrict__ w_gen,
                             const float* __restrict__ b_ih, const float* __restrict__ b_hh,
                             unsigned short* __restrict__ w_ihK_bf, unsigned short* __restrict__ w_hh_bf,
                             unsigned short* __restrict__ w_h2h_bf, unsigned short* __restrict__ w_i2h_bf,
                             unsigned short* __restrict__ w_gen_bf, float* __restrict__ biasc) {
  int idx = blockIdx.x * 256 + threadIdx.x;       // 0 .. 262143
  int n = idx >> 8, k = idx & 255;
  w_ihK_bf[idx] = f2bf(w_ih[(size_t)n * KX_ + k]);   // first 256 cols only
  w_hh_bf[idx]  = f2bf(w_hh[idx]);
  if (n < 256) { w_h2h_bf[idx] = f2bf(w_h2h[idx]); w_i2h_bf[idx] = f2bf(w_i2h[idx]); }
  if (n < 48)  w_gen_bf[idx] = (n < C_) ? f2bf(w_gen[(size_t)n * 256 + k]) : (unsigned short)0;
  if (k == 0)  biasc[n] = b_ih[n] + b_hh[n];
}

// enc fp32 -> bf16. grid: 32768 blocks x 256.
__global__ void conv_enc(const float* __restrict__ enc, unsigned short* __restrict__ enc_bf) {
  size_t idx = (size_t)blockIdx.x * 256 + threadIdx.x;
  enc_bf[idx] = f2bf(enc[idx]);
}

// H_proj[m,h] = sum_d enc[m,d]*w_i2h[h,d]; M=B*T=32768, N=256, K=256.
// One 16x16 tile per wave; 8 waves/block; 4096 blocks.
__global__ void __launch_bounds__(256)
hproj_gemm(const unsigned short* __restrict__ enc_bf, const unsigned short* __restrict__ w_bf,
           float* __restrict__ Hproj) {
  int lane = threadIdx.x & 31;
  int wave = threadIdx.x >> 5;
  int tile = blockIdx.x * 8 + wave;              // 0..32767
  int mtile = tile >> 4, ntile = tile & 15;
  v8f acc = v8zero();
  const unsigned short* Abase = enc_bf + (size_t)mtile * 16 * 256;
#pragma unroll
  for (int kb = 0; kb < 8; ++kb) {
    v16bf a = load_frag_A(Abase, lane, kb * 32);
    v16bf b = load_frag_B(w_bf, lane, ntile, kb * 32);
    acc = wmma_bf16(a, b, acc);
  }
  int col = lane & 15, hi = lane >> 4;
#pragma unroll
  for (int r = 0; r < 8; ++r) {
    int m = mtile * 16 + r + 8 * hi;
    Hproj[(size_t)m * 256 + ntile * 16 + col] = acc[r];
  }
}

// ---------------- fused recurrent decode ----------------
// 64 blocks x 256 threads; block handles 16 batch rows through all 26 steps.
__global__ void __launch_bounds__(256)
decode_kernel(const float* __restrict__ enc, const int* __restrict__ text,
              const float* __restrict__ w_score, const float* __restrict__ b_h2h,
              const float* __restrict__ w_ih_full, const float* __restrict__ b_gen,
              const unsigned short* __restrict__ w_h2h_bf, const unsigned short* __restrict__ w_ihK_bf,
              const unsigned short* __restrict__ w_hh_bf, const unsigned short* __restrict__ w_gen_bf,
              const float* __restrict__ biasc, const float* __restrict__ Hproj,
              float* __restrict__ out) {
  __shared__ float s_c[16 * 256];                         // cell state (fp32)
  __shared__ float s_hp[16 * 256];                        // h @ w_h2h^T + b
  __shared__ float s_e[16 * 32];
  __shared__ float s_alpha[16 * 32];
  __shared__ float s_ws[256];
  __shared__ float s_biasc[1024];
  __shared__ __align__(16) unsigned short s_hbf[16 * 256];    // h (bf16) WMMA A
  __shared__ __align__(16) unsigned short s_ctxbf[16 * 256];  // context (bf16) WMMA A
  __shared__ int s_cls[16];

  const int tid  = threadIdx.x;
  const int lane = tid & 31;
  const int wave = tid >> 5;
  const int b0   = blockIdx.x * 16;
  const int col  = lane & 15;
  const int hi   = lane >> 4;

  for (int i = tid; i < 16 * 256; i += 256) { s_c[i] = 0.f; s_hbf[i] = 0; }
  if (tid < 256) s_ws[tid] = w_score[tid];
  for (int i = tid; i < 1024; i += 256) s_biasc[i] = biasc[i];
  __syncthreads();

  for (int step = 0; step < STEPS_; ++step) {
    // ---- P1: hp = h @ w_h2h^T + b_h2h   (16 tiles, 2 per wave) ----
    {
      v8f a0 = v8zero(), a1 = v8zero();
      int n0 = wave * 2, n1 = wave * 2 + 1;
#pragma unroll
      for (int kb = 0; kb < 8; ++kb) {
        v16bf a = load_frag_A(s_hbf, lane, kb * 32);
        a0 = wmma_bf16(a, load_frag_B(w_h2h_bf, lane, n0, kb * 32), a0);
        a1 = wmma_bf16(a, load_frag_B(w_h2h_bf, lane, n1, kb * 32), a1);
      }
#pragma unroll
      for (int r = 0; r < 8; ++r) {
        int row = r + 8 * hi;
        int cA = n0 * 16 + col, cB = n1 * 16 + col;
        s_hp[row * 256 + cA] = a0[r] + b_h2h[cA];
        s_hp[row * 256 + cB] = a1[r] + b_h2h[cB];
      }
      if (tid < 16) s_cls[tid] = text[(size_t)(b0 + tid) * STEPS_ + step];
    }
    __syncthreads();

    // ---- P2: e[row,t] = sum_h tanh(Hproj + hp) * ws  (16-lane cooperative) ----
    {
      int g = tid >> 4, l16 = tid & 15;
      for (int p = g; p < 16 * 32; p += 16) {
        int row = p >> 5, t = p & 31;
        const float* hpr = s_hp + row * 256;
        const float* Hr  = Hproj + ((size_t)(b0 + row) * T_ + t) * 256;
        float acc = 0.f;
#pragma unroll
        for (int k = 0; k < 16; ++k) {
          int hh = l16 + 16 * k;
          acc += tanhf(Hr[hh] + hpr[hh]) * s_ws[hh];
        }
        for (int off = 8; off; off >>= 1) acc += __shfl_xor(acc, off, 16);
        if (l16 == 0) s_e[row * 32 + t] = acc;
      }
    }
    __syncthreads();
    if (tid < 16) {                                     // softmax over T=32
      int row = tid;
      float m = -1e30f;
      for (int t = 0; t < 32; ++t) m = fmaxf(m, s_e[row * 32 + t]);
      float s = 0.f;
      for (int t = 0; t < 32; ++t) { float v = __expf(s_e[row * 32 + t] - m); s_alpha[row * 32 + t] = v; s += v; }
      float inv = 1.f / s;
      for (int t = 0; t < 32; ++t) s_alpha[row * 32 + t] *= inv;
    }
    __syncthreads();

    // ---- P3: context[row,d] = sum_t alpha * enc ----
    {
      int row = tid >> 4, j = tid & 15;
      float acc[16];
#pragma unroll
      for (int k = 0; k < 16; ++k) acc[k] = 0.f;
      const float* erow = enc + (size_t)(b0 + row) * T_ * 256;
      for (int t = 0; t < 32; ++t) {
        float a = s_alpha[row * 32 + t];
        const float* ep = erow + t * 256;
#pragma unroll
        for (int k = 0; k < 16; ++k) acc[k] += a * ep[j + 16 * k];
      }
#pragma unroll
      for (int k = 0; k < 16; ++k) s_ctxbf[row * 256 + j + 16 * k] = f2bf(acc[k]);
    }
    __syncthreads();

    // ---- P4: gates = ctx@w_ihK^T + h@w_hh^T ; wave w owns i/f/g/o tiles of
    //          the same 32 hidden cols (tiles w, w+8, w+16, ..., w+56) ----
    v8f acc[4][2];
#pragma unroll
    for (int gg = 0; gg < 4; ++gg)
#pragma unroll
      for (int hf = 0; hf < 2; ++hf) acc[gg][hf] = v8zero();
#pragma unroll
    for (int kb = 0; kb < 8; ++kb) {
      v16bf aC = load_frag_A(s_ctxbf, lane, kb * 32);
      v16bf aH = load_frag_A(s_hbf,   lane, kb * 32);
#pragma unroll
      for (int gg = 0; gg < 4; ++gg)
#pragma unroll
        for (int hf = 0; hf < 2; ++hf) {
          int nt = wave + 8 * hf + 16 * gg;
          acc[gg][hf] = wmma_bf16(aC, load_frag_B(w_ihK_bf, lane, nt, kb * 32), acc[gg][hf]);
          acc[gg][hf] = wmma_bf16(aH, load_frag_B(w_hh_bf,  lane, nt, kb * 32), acc[gg][hf]);
        }
    }
    __syncthreads();   // all waves finished READING s_hbf/s_ctxbf

    // ---- P5: LSTM elementwise directly on WMMA accumulator layout ----
#pragma unroll
    for (int hf = 0; hf < 2; ++hf) {
      int cb = (wave + 8 * hf) * 16 + col;      // hidden col 0..255
#pragma unroll
      for (int r = 0; r < 8; ++r) {
        int row = r + 8 * hi;
        int cls = s_cls[row];
        const float* wc = w_ih_full + IN_ + cls;            // one-hot column add
        float gi = acc[0][hf][r] + s_biasc[cb]        + wc[(size_t)(cb)        * KX_];
        float gf = acc[1][hf][r] + s_biasc[256 + cb]  + wc[(size_t)(256 + cb)  * KX_];
        float gG = acc[2][hf][r] + s_biasc[512 + cb]  + wc[(size_t)(512 + cb)  * KX_];
        float go = acc[3][hf][r] + s_biasc[768 + cb]  + wc[(size_t)(768 + cb)  * KX_];
        float cn = sigm(gf) * s_c[row * 256 + cb] + sigm(gi) * tanhf(gG);
        float hn = sigm(go) * tanhf(cn);
        s_c[row * 256 + cb]   = cn;
        s_hbf[row * 256 + cb] = f2bf(hn);
      }
    }
    __syncthreads();

    // ---- P6: probs = h @ w_gen^T + b_gen  (N padded to 48 -> 3 tiles) ----
    if (wave < 3) {
      v8f pa = v8zero();
#pragma unroll
      for (int kb = 0; kb < 8; ++kb) {
        v16bf a = load_frag_A(s_hbf, lane, kb * 32);
        pa = wmma_bf16(a, load_frag_B(w_gen_bf, lane, wave, kb * 32), pa);
      }
      int n = wave * 16 + col;
      if (n < C_) {
        float bg = b_gen[n];
#pragma unroll
        for (int r = 0; r < 8; ++r) {
          int row = r + 8 * hi;
          out[((size_t)(b0 + row) * STEPS_ + step) * C_ + n] = pa[r] + bg;
        }
      }
    }
    __syncthreads();
  }
}

// ---------------- launch ----------------
extern "C" void kernel_launch(void* const* d_in, const int* in_sizes, int n_in,
                              void* d_out, int out_size, void* d_ws, size_t ws_size,
                              hipStream_t stream) {
  (void)in_sizes; (void)n_in; (void)out_size; (void)ws_size;
  const float* enc     = (const float*)d_in[0];
  const int*   text    = (const int*)d_in[1];
  // d_in[2] semantics, d_in[3] is_train: unused (as in reference)
  const float* w_i2h   = (const float*)d_in[4];
  const float* w_h2h   = (const float*)d_in[5];
  const float* b_h2h   = (const float*)d_in[6];
  const float* w_score = (const float*)d_in[7];
  const float* w_ih    = (const float*)d_in[8];
  const float* w_hh    = (const float*)d_in[9];
  const float* b_ih    = (const float*)d_in[10];
  const float* b_hh    = (const float*)d_in[11];
  const float* w_gen   = (const float*)d_in[12];
  const float* b_gen   = (const float*)d_in[13];

  char* ws = (char*)d_ws;
  unsigned short* enc_bf   = (unsigned short*)(ws);                          // 16.78 MB
  float*          Hproj    = (float*)(ws + 16777216);                        // 33.55 MB
  unsigned short* w_i2h_bf = (unsigned short*)(ws + 50331648);               // 128 KB
  unsigned short* w_h2h_bf = (unsigned short*)(ws + 50331648 + 131072);      // 128 KB
  unsigned short* w_ihK_bf = (unsigned short*)(ws + 50331648 + 262144);      // 512 KB
  unsigned short* w_hh_bf  = (unsigned short*)(ws + 50331648 + 786432);      // 512 KB
  unsigned short* w_gen_bf = (unsigned short*)(ws + 50331648 + 1310720);     // 24 KB
  float*          biasc    = (float*)(ws + 50331648 + 1335296);              // 4 KB

  conv_weights<<<1024, 256, 0, stream>>>(w_ih, w_hh, w_h2h, w_i2h, w_gen, b_ih, b_hh,
                                         w_ihK_bf, w_hh_bf, w_h2h_bf, w_i2h_bf,
                                         w_gen_bf, biasc);
  conv_enc<<<(B_ * T_ * IN_) / 256, 256, 0, stream>>>(enc, enc_bf);
  hproj_gemm<<<(B_ * T_ / 16) * (H_ / 16) / 8, 256, 0, stream>>>(enc_bf, w_i2h_bf, Hproj);
  decode_kernel<<<B_ / 16, 256, 0, stream>>>(enc, text, w_score, b_h2h, w_ih, b_gen,
                                             w_h2h_bf, w_ihK_bf, w_hh_bf, w_gen_bf,
                                             biasc, Hproj, (float*)d_out);
}
